// RnnBlock_80298708566129
// MI455X (gfx1250) — matrix operand order repaired and checked by reading, not compiled, not run on hardware
//
#include <hip/hip_runtime.h>
#include <hip/hip_bf16.h>
#include <string.h>

// ---------------------------------------------------------------------------
// Problem constants (from reference): B=16, T=400, F=512, U=1024
// ---------------------------------------------------------------------------
#define BB   16
#define TT   400
#define FF   512
#define UU   1024
#define G4   (4 * UU)        // 4096 gate columns per direction
#define MROWS (BB * TT)      // 6400
#define NBLK_LSTM 64         // 32 chunks per direction * 2 directions
#define EPS  1e-3f

typedef __attribute__((ext_vector_type(16))) __bf16 v16bf;
typedef __attribute__((ext_vector_type(8)))  float  v8f;
typedef int v4i_gcc __attribute__((vector_size(16)));   // matches builtin proto

union Frag32B { uint4 q[2]; v16bf v; };

// CDNA5 async global->LDS path (ASYNCcnt), if toolchain exposes the builtins.
#if defined(__has_builtin)
#if __has_builtin(__builtin_amdgcn_global_load_async_to_lds_b128) && \
    __has_builtin(__builtin_amdgcn_s_wait_asynccnt)
#define HAVE_ASYNC_LDS 1
#endif
#endif

// ---------------------------------------------------------------------------
// Workspace layout (bytes)
// ---------------------------------------------------------------------------
#define OFF_XBF  ((size_t)0)                         // x bf16:  6400*512*2           = 6,553,600
#define OFF_KT   ((size_t)6553600)                   // kT bf16 [2][4096][512]        = 8,388,608
#define OFF_RT   ((size_t)14942208)                  // rT bf16 [2][4096][1024]       = 16,777,216
#define OFF_XW   ((size_t)31719424)                  // xw f32  [2][6400][4096]       = 209,715,200
#define OFF_H    ((size_t)241434624)                 // h bf16  [2 dir][2 buf][16][1024]
#define OFF_BAR  ((size_t)241565696)                 // barrier counter

// ---------------------------------------------------------------------------
// Helpers
// ---------------------------------------------------------------------------
__device__ __forceinline__ unsigned short f2bf(float f) {
    unsigned u = __float_as_uint(f);
    u += 0x7FFFu + ((u >> 16) & 1u);          // round-to-nearest-even
    return (unsigned short)(u >> 16);
}

__device__ __forceinline__ float sigf(float x) {
    return 1.0f / (1.0f + __expf(-x));
}

// 16-bit A fragment (16x32, row-major source, documented lane striping):
//   lanes 0-15 : row M=lane,  K = {k0..k0+7,  k0+16..k0+23}
//   lanes 16-31: row M=l-16,  K = {k0+8..k0+15, k0+24..k0+31}
__device__ __forceinline__ v16bf load_frag_a(const unsigned short* base, int ld,
                                             int m16, int k0, int lane) {
    int r  = lane & 15;
    int ko = (lane >= 16) ? 8 : 0;
    const unsigned short* p = base + (size_t)(m16 + r) * ld + k0 + ko;
    Frag32B f;
    f.q[0] = *(const uint4*)(p);
    f.q[1] = *(const uint4*)(p + 16);
    return f.v;
}

// 16-bit B fragment (32x16) from *transposed* [N][K] storage:
//   lanes 0-15 : col N=lane,  K = {k0..k0+7,  k0+16..k0+23}
//   lanes 16-31: col N=l-16,  K = {k0+8..k0+15, k0+24..k0+31}
__device__ __forceinline__ v16bf load_frag_bt(const unsigned short* base, int ld,
                                              int n16, int k0, int lane) {
    int n  = lane & 15;
    int ko = (lane >= 16) ? 8 : 0;
    const unsigned short* p = base + (size_t)(n16 + n) * ld + k0 + ko;
    Frag32B f;
    f.q[0] = *(const uint4*)(p);
    f.q[1] = *(const uint4*)(p + 16);
    return f.v;
}

// LDS variants (emit ds_load_b128)
__device__ __forceinline__ v16bf load_frag_a_lds(const unsigned short* base, int ld,
                                                 int k0, int lane) {
    int r  = lane & 15;
    int ko = (lane >= 16) ? 8 : 0;
    const unsigned short* p = base + r * ld + k0 + ko;
    Frag32B f;
    f.q[0] = *(const uint4*)(p);
    f.q[1] = *(const uint4*)(p + 16);
    return f.v;
}

__device__ __forceinline__ v16bf load_frag_bt_lds(const unsigned short* base, int ld,
                                                  int n16, int k0, int lane) {
    int n  = lane & 15;
    int ko = (lane >= 16) ? 8 : 0;
    const unsigned short* p = base + (n16 + n) * ld + k0 + ko;
    Frag32B f;
    f.q[0] = *(const uint4*)(p);
    f.q[1] = *(const uint4*)(p + 16);
    return f.v;
}

__device__ __forceinline__ v8f wmma_bf16(v16bf a, v16bf b, v8f c) {
    return __builtin_amdgcn_wmma_f32_16x16x32_bf16(false, a, false, b,
                                                   (short)0, c, false, false);
}

// ---------------------------------------------------------------------------
// Kernel 0: per-launch workspace init (h buffers + grid barrier)
// ---------------------------------------------------------------------------
__global__ void init_ws_kernel(unsigned short* h, unsigned* bar) {
    int i = blockIdx.x * blockDim.x + threadIdx.x;
    if (i < 2 * 2 * BB * UU) h[i] = 0;
    if (i == 0) bar[0] = 0u;
}

// ---------------------------------------------------------------------------
// Kernel 1: f32 -> bf16 elementwise (x)
// ---------------------------------------------------------------------------
__global__ void cvt_kernel(const float* src, unsigned short* dst, int n) {
    int i = blockIdx.x * blockDim.x + threadIdx.x;
    if (i < n) dst[i] = f2bf(src[i]);
}

// ---------------------------------------------------------------------------
// Kernel 2: f32 [rows][cols] -> bf16 transposed [cols][rows]
// ---------------------------------------------------------------------------
__global__ void cvtT_kernel(const float* src, unsigned short* dst, int rows, int cols) {
    int i = blockIdx.x * blockDim.x + threadIdx.x;
    int total = rows * cols;
    if (i >= total) return;
    int c = i / rows;       // dst-major: write coalesced in dst
    int r = i - c * rows;
    dst[(size_t)c * rows + r] = f2bf(src[(size_t)r * cols + c]);
}

// ---------------------------------------------------------------------------
// Kernel 3: input-projection GEMM  xw[d] = x_bf16 @ kT[d]^T   (f32 accumulate)
//   grid: (MROWS/64, G4/256, 2), block: 256 threads (8 waves)
//   wave (mw,nw) computes a 32x64 sub-tile = 8 WMMA tiles (6 frag loads / 8 wmma)
// ---------------------------------------------------------------------------
__global__ void gemm_xw_kernel(const unsigned short* __restrict__ xbf,
                               const unsigned short* __restrict__ kT,
                               float* __restrict__ xw) {
    const int d = blockIdx.z;
    const unsigned short* Bt = kT + (size_t)d * G4 * FF;   // [4096][512]
    float* C = xw + (size_t)d * MROWS * G4;

    const int lane = threadIdx.x & 31;
    const int w    = threadIdx.x >> 5;
    const int mw   = w >> 2;              // 0..1
    const int nw   = w & 3;               // 0..3
    const int m0   = blockIdx.x * 64 + mw * 32;
    const int n0   = blockIdx.y * 256 + nw * 64;

    v8f acc[2][4];
    for (int i = 0; i < 2; ++i)
        for (int j = 0; j < 4; ++j)
            for (int e = 0; e < 8; ++e) acc[i][j][e] = 0.0f;

    for (int k0 = 0; k0 < FF; k0 += 32) {
        v16bf a0 = load_frag_a(xbf, FF, m0,      k0, lane);
        v16bf a1 = load_frag_a(xbf, FF, m0 + 16, k0, lane);
        v16bf b0 = load_frag_bt(Bt, FF, n0,      k0, lane);
        v16bf b1 = load_frag_bt(Bt, FF, n0 + 16, k0, lane);
        v16bf b2 = load_frag_bt(Bt, FF, n0 + 32, k0, lane);
        v16bf b3 = load_frag_bt(Bt, FF, n0 + 48, k0, lane);
        acc[0][0] = wmma_bf16(a0, b0, acc[0][0]);
        acc[0][1] = wmma_bf16(a0, b1, acc[0][1]);
        acc[0][2] = wmma_bf16(a0, b2, acc[0][2]);
        acc[0][3] = wmma_bf16(a0, b3, acc[0][3]);
        acc[1][0] = wmma_bf16(a1, b0, acc[1][0]);
        acc[1][1] = wmma_bf16(a1, b1, acc[1][1]);
        acc[1][2] = wmma_bf16(a1, b2, acc[1][2]);
        acc[1][3] = wmma_bf16(a1, b3, acc[1][3]);
    }

    // C/D layout: VGPR e -> (M = e (+8 if lane>=16), N = lane&15)
    const int mofs = (lane >= 16) ? 8 : 0;
    const int nn   = lane & 15;
    for (int i = 0; i < 2; ++i)
        for (int j = 0; j < 4; ++j)
            for (int e = 0; e < 8; ++e) {
                int m = m0 + i * 16 + e + mofs;
                int n = n0 + j * 16 + nn;
                C[(size_t)m * G4 + n] = acc[i][j][e];
            }
}

// ---------------------------------------------------------------------------
// Grid-wide barrier: monotone counter, re-zeroed each launch by init kernel
// ---------------------------------------------------------------------------
__device__ __forceinline__ void grid_sync(unsigned* bar, unsigned phase) {
    __syncthreads();
    if (threadIdx.x == 0) {
        __threadfence();
        atomicAdd(bar, 1u);
        const unsigned target = phase * (unsigned)NBLK_LSTM;
        while (atomicAdd(bar, 0u) < target) {
            __builtin_amdgcn_s_sleep(1);
        }
    }
    __syncthreads();
    __threadfence();   // acquire: invalidate near caches before re-reading h
}

// ---------------------------------------------------------------------------
// Kernel 4: persistent bidirectional LSTM recurrence + fused BatchNorm.
//   64 workgroups: blockIdx >> 5 = direction, blockIdx & 31 = unit chunk q.
//   Each WG owns units u in [32q, 32q+32) => gate columns {u, u+1024, u+2048,
//   u+3072}. Its 128x1024 bf16 slice of rT (256 KB) is preloaded into LDS and
//   reused for all 400 steps. Each step, the 32 KB broadcast h matrix is
//   staged into LDS (async-to-LDS when available), then wave w (gate = w>>1,
//   half = w&1) runs one 16x16 WMMA accumulator over K=1024
//   (32 x v_wmma_f32_16x16x32_bf16 per step, operands entirely from LDS).
//   Cell state c lives in 2 registers per thread for the whole sequence.
//   LDS: 256 KB weights + 8 KB z-buffer + 32 KB h-stage = 296 KB (< 320 KB).
// ---------------------------------------------------------------------------
__global__ void lstm_persistent_kernel(const float* __restrict__ xw,
                                       const unsigned short* __restrict__ rT,
                                       const float* __restrict__ b_fwd,
                                       const float* __restrict__ b_bwd,
                                       const float* __restrict__ gamma,
                                       const float* __restrict__ beta,
                                       const float* __restrict__ mmean,
                                       const float* __restrict__ mvar,
                                       unsigned short* __restrict__ hbuf,
                                       float* __restrict__ out,
                                       unsigned* __restrict__ bar) {
    extern __shared__ char smem[];
    unsigned short* rW     = (unsigned short*)smem;                      // 256 KB
    float*          zbf    = (float*)(smem + 128 * UU * 2);              // 8 KB
    unsigned short* hstage = (unsigned short*)(smem + 128 * UU * 2 + 16 * 128 * 4); // 32 KB

    const int d    = blockIdx.x >> 5;
    const int q    = blockIdx.x & 31;
    const int u0   = q * 32;
    const int tid  = threadIdx.x;
    const int lane = tid & 31;
    const int w    = tid >> 5;
    const int gate = w >> 1;
    const int half = w & 1;

    // ---- Preload the 128x1024 bf16 weight slice into LDS (once) ----------
    // local row lr = g*32 + j  <->  global rT row g*1024 + u0 + j
    const unsigned short* rTd = rT + (size_t)d * G4 * UU;
    for (int idx = tid; idx < 128 * (UU / 16); idx += 256) {   // uint4 units
        int row   = idx >> 6;          // /64 uint4 per row
        int col16 = idx & 63;
        int gg = row >> 5, j = row & 31;
        ((uint4*)rW)[idx] =
            *(const uint4*)(rTd + ((size_t)(gg * UU + u0 + j)) * UU + col16 * 16);
    }

    // ---- Per-thread elementwise mapping: 2 cells (b, u_local) ------------
    const int e0  = tid * 2;           // element ids 0..511 over [16 b][32 u]
    const int b_  = e0 >> 5;
    const int ul0 = e0 & 31;
    const int ul1 = ul0 + 1;
    const int cg0 = d * UU + u0 + ul0;
    const int cg1 = d * UU + u0 + ul1;
    const float inv0 = gamma[cg0] * rsqrtf(mvar[cg0] + EPS);
    const float inv1 = gamma[cg1] * rsqrtf(mvar[cg1] + EPS);
    const float add0 = beta[cg0] - mmean[cg0] * inv0;
    const float add1 = beta[cg1] - mmean[cg1] * inv1;
    const float* bias = (d == 0) ? b_fwd : b_bwd;
    const float bi0 = bias[0 * UU + u0 + ul0], bi1 = bias[0 * UU + u0 + ul1];
    const float bf0 = bias[1 * UU + u0 + ul0], bf1 = bias[1 * UU + u0 + ul1];
    const float bg0 = bias[2 * UU + u0 + ul0], bg1 = bias[2 * UU + u0 + ul1];
    const float bo0 = bias[3 * UU + u0 + ul0], bo1 = bias[3 * UU + u0 + ul1];

    float c0 = 0.0f, c1 = 0.0f;

    unsigned short* h0 = hbuf + (size_t)d * 2 * BB * UU;
    unsigned short* h1 = h0 + BB * UU;

    const float* xwd = xw + (size_t)d * MROWS * G4;
    const int lrb = gate * 32 + half * 16;   // base LDS row of this wave's B tile

    __syncthreads();                         // rW preload visible

    for (int t = 0; t < TT; ++t) {
        const unsigned short* hc = (t & 1) ? h1 : h0;
        unsigned short*       hn = (t & 1) ? h0 : h1;

        // ---- stage broadcast h [16][1024] bf16 into LDS (32 KB) ----------
        {
            const uint4* hcq = (const uint4*)hc;
            uint4*       hsq = (uint4*)hstage;
            for (int idx = tid; idx < BB * UU / 8; idx += 256) {
#if defined(HAVE_ASYNC_LDS)
                __builtin_amdgcn_global_load_async_to_lds_b128(
                    (__attribute__((address_space(1))) v4i_gcc*)(hcq + idx),
                    (__attribute__((address_space(3))) v4i_gcc*)(hsq + idx),
                    0, 0);
#else
                hsq[idx] = hcq[idx];
#endif
            }
#if defined(HAVE_ASYNC_LDS)
            __builtin_amdgcn_s_wait_asynccnt(0);
#endif
        }
        __syncthreads();

        // ---- z_tile = h @ r  (one 16x16 tile per wave, K = 1024, all LDS) -
        v8f acc;
        for (int e = 0; e < 8; ++e) acc[e] = 0.0f;
        for (int k0 = 0; k0 < UU; k0 += 32) {
            v16bf a = load_frag_a_lds(hstage, UU, k0, lane);       // h (LDS)
            v16bf b = load_frag_bt_lds(rW, UU, lrb, k0, lane);     // weights (LDS)
            acc = wmma_bf16(a, b, acc);
        }

        // ---- scatter accumulator into z-buffer ---------------------------
        {
            const int mofs = (lane >= 16) ? 8 : 0;
            const int colb = gate * 32 + half * 16 + (lane & 15);
            for (int e = 0; e < 8; ++e)
                zbf[(e + mofs) * 128 + colb] = acc[e];
        }
        __syncthreads();

        // ---- cell update + h + fused BatchNorm ---------------------------
        const int tx = (d == 0) ? t : (TT - 1 - t);
        const float* xwrow = xwd + ((size_t)b_ * TT + tx) * G4;
        const float* zrow  = zbf + b_ * 128;

        {   // element 0
            float zi = zrow[0 * 32 + ul0] + xwrow[0 * UU + u0 + ul0] + bi0;
            float zf = zrow[1 * 32 + ul0] + xwrow[1 * UU + u0 + ul0] + bf0;
            float zg = zrow[2 * 32 + ul0] + xwrow[2 * UU + u0 + ul0] + bg0;
            float zo = zrow[3 * 32 + ul0] + xwrow[3 * UU + u0 + ul0] + bo0;
            c0 = sigf(zf) * c0 + sigf(zi) * tanhf(zg);
            float h = sigf(zo) * tanhf(c0);
            hn[(size_t)b_ * UU + u0 + ul0] = f2bf(h);
            out[((size_t)b_ * TT + tx) * (2 * UU) + cg0] = h * inv0 + add0;
        }
        {   // element 1
            float zi = zrow[0 * 32 + ul1] + xwrow[0 * UU + u0 + ul1] + bi1;
            float zf = zrow[1 * 32 + ul1] + xwrow[1 * UU + u0 + ul1] + bf1;
            float zg = zrow[2 * 32 + ul1] + xwrow[2 * UU + u0 + ul1] + bg1;
            float zo = zrow[3 * 32 + ul1] + xwrow[3 * UU + u0 + ul1] + bo1;
            c1 = sigf(zf) * c1 + sigf(zi) * tanhf(zg);
            float h = sigf(zo) * tanhf(c1);
            hn[(size_t)b_ * UU + u0 + ul1] = f2bf(h);
            out[((size_t)b_ * TT + tx) * (2 * UU) + cg1] = h * inv1 + add1;
        }

        // ---- prefetch next step's xw row (global_prefetch_b8) ------------
        if (t + 1 < TT) {
            const int txn = (d == 0) ? (t + 1) : (TT - 2 - t);
            const float* nxt = xwd + ((size_t)b_ * TT + txn) * G4;
            __builtin_prefetch(nxt + 0 * UU + u0 + ul0, 0, 0);
            __builtin_prefetch(nxt + 1 * UU + u0 + ul0, 0, 0);
            __builtin_prefetch(nxt + 2 * UU + u0 + ul0, 0, 0);
            __builtin_prefetch(nxt + 3 * UU + u0 + ul0, 0, 0);
        }

        grid_sync(bar, (unsigned)(t + 1));   // includes syncthreads (zbf/hstage reuse)
    }
}

// ---------------------------------------------------------------------------
// Host launcher
// ---------------------------------------------------------------------------
extern "C" void kernel_launch(void* const* d_in, const int* in_sizes, int n_in,
                              void* d_out, int out_size, void* d_ws, size_t ws_size,
                              hipStream_t stream) {
    (void)in_sizes; (void)n_in; (void)out_size; (void)ws_size;

    const float* x      = (const float*)d_in[0];
    const float* k_fwd  = (const float*)d_in[1];
    const float* r_fwd  = (const float*)d_in[2];
    const float* b_fwd  = (const float*)d_in[3];
    const float* k_bwd  = (const float*)d_in[4];
    const float* r_bwd  = (const float*)d_in[5];
    const float* b_bwd  = (const float*)d_in[6];
    const float* gamma  = (const float*)d_in[7];
    const float* beta   = (const float*)d_in[8];
    const float* mmean  = (const float*)d_in[9];
    const float* mvar   = (const float*)d_in[10];
    float* out = (float*)d_out;

    char* ws = (char*)d_ws;
    unsigned short* xbf  = (unsigned short*)(ws + OFF_XBF);
    unsigned short* kT   = (unsigned short*)(ws + OFF_KT);
    unsigned short* rT   = (unsigned short*)(ws + OFF_RT);
    float*          xw   = (float*)(ws + OFF_XW);
    unsigned short* hbuf = (unsigned short*)(ws + OFF_H);
    unsigned*       bar  = (unsigned*)(ws + OFF_BAR);

    // 0) init h state + barrier
    init_ws_kernel<<<(2 * 2 * BB * UU + 255) / 256, 256, 0, stream>>>(hbuf, bar);

    // 1) x -> bf16
    {
        int n = BB * TT * FF;
        cvt_kernel<<<(n + 255) / 256, 256, 0, stream>>>(x, xbf, n);
    }

    // 2) transposed bf16 weights: kT[d] = k[d]^T, rT[d] = r[d]^T
    {
        int nk = FF * G4;
        cvtT_kernel<<<(nk + 255) / 256, 256, 0, stream>>>(k_fwd, kT, FF, G4);
        cvtT_kernel<<<(nk + 255) / 256, 256, 0, stream>>>(k_bwd, kT + (size_t)G4 * FF, FF, G4);
        int nr = UU * G4;
        cvtT_kernel<<<(nr + 255) / 256, 256, 0, stream>>>(r_fwd, rT, UU, G4);
        cvtT_kernel<<<(nr + 255) / 256, 256, 0, stream>>>(r_bwd, rT + (size_t)G4 * UU, UU, G4);
    }

    // 3) input-projection GEMM (both directions)
    {
        dim3 grid(MROWS / 64, G4 / 256, 2);
        gemm_xw_kernel<<<grid, 256, 0, stream>>>(xbf, kT, xw);
    }

    // 4) persistent recurrence + fused BN
    {
        const size_t smem = 128 * UU * 2      // weights
                          + 16 * 128 * 4      // z-buffer
                          + BB * UU * 2;      // h stage
        (void)hipFuncSetAttribute((const void*)lstm_persistent_kernel,
                                  hipFuncAttributeMaxDynamicSharedMemorySize,
                                  (int)smem);
        lstm_persistent_kernel<<<NBLK_LSTM, 256, smem, stream>>>(
            xw, rT, b_fwd, b_bwd, gamma, beta, mmean, mvar, hbuf, out, bar);
    }
}
